// RNN_21809843929473
// MI455X (gfx1250) — matrix operand order repaired and checked by reading, not compiled, not run on hardware
//
#include <hip/hip_runtime.h>

// CDNA5 / gfx1250 fused 2-layer Elman RNN.
// Recurrent state lives in the WMMA *B* operand (K = augmented hidden, N = batch),
// weights live in the resident *A* operand. Hidden units are relabeled with the
// permutation slot(t7,r) = 32*(t7>>1) + 16*(r>=8) + 8*(t7&1) + (r&7) so that the
// D -> next-step-B conversion is a purely lane-local f32->f16 repack:
// no LDS, no shuffles, no selects in the steady loop.

typedef __attribute__((ext_vector_type(16))) _Float16 v16h;
typedef __attribute__((ext_vector_type(8)))  float    v8f;

union V16H { v16h v; _Float16 h[16]; unsigned u[8]; };
union V8F  { v8f  v; float f[8]; };

// Batched CDNA5 hardware tanh (transcendental VALU). Consecutive independent
// v_tanh_f32 ops satisfy the "1 independent op after trans" rule for each
// other; a single trailing v_nop covers the last one.
__device__ __forceinline__ void tanh8(V8F& a) {
  asm("v_tanh_f32 %0, %0\n\t"
      "v_tanh_f32 %1, %1\n\t"
      "v_tanh_f32 %2, %2\n\t"
      "v_tanh_f32 %3, %3\n\t"
      "v_tanh_f32 %4, %4\n\t"
      "v_tanh_f32 %5, %5\n\t"
      "v_tanh_f32 %6, %6\n\t"
      "v_tanh_f32 %7, %7\n\t"
      "v_nop"
      : "+v"(a.f[0]), "+v"(a.f[1]), "+v"(a.f[2]), "+v"(a.f[3]),
        "+v"(a.f[4]), "+v"(a.f[5]), "+v"(a.f[6]), "+v"(a.f[7]));
}

__device__ __forceinline__ void tanh5(float& a0, float& a1, float& a2,
                                      float& a3, float& a4) {
  asm("v_tanh_f32 %0, %0\n\t"
      "v_tanh_f32 %1, %1\n\t"
      "v_tanh_f32 %2, %2\n\t"
      "v_tanh_f32 %3, %3\n\t"
      "v_tanh_f32 %4, %4\n\t"
      "v_nop"
      : "+v"(a0), "+v"(a1), "+v"(a2), "+v"(a3), "+v"(a4));
}

// pack two f32 into one dword of two f16 (v_cvt_pk_rtz_f16_f32)
__device__ __forceinline__ unsigned pk2(float lo, float hi) {
  auto p = __builtin_amdgcn_cvt_pkrtz(lo, hi);
  return __builtin_bit_cast(unsigned, p);
}

__global__ __launch_bounds__(256) void rnn_fused(
    const float* __restrict__ x,
    const float* __restrict__ Wih1, const float* __restrict__ Whh1,
    const float* __restrict__ bih1, const float* __restrict__ bhh1,
    const float* __restrict__ Wih2, const float* __restrict__ Whh2,
    const float* __restrict__ bih2, const float* __restrict__ bhh2,
    const float* __restrict__ W3,   const float* __restrict__ b3,
    float* __restrict__ out, int Btot)
{
  constexpr int T = 256, H2 = 100;
  const int lane = threadIdx.x & 31;
  const int wave = threadIdx.x >> 5;
  const int gwave = blockIdx.x * (blockDim.x >> 5) + wave;
  const int base = gwave * 16;          // 16 batch rows per wave
  if (base >= Btot) return;
  const int col = lane & 15;            // batch column within tile (B/C/D lane)
  const int hs  = lane >> 4;            // lane half (0: lanes 0-15, 1: 16-31)

  // ---- layer-1 parameters (tiny; uniform -> compiler scalarizes)
  float wih1[5], b1c[5], whh1[5][5];
  #pragma unroll
  for (int i = 0; i < 5; ++i) { wih1[i] = Wih1[i]; b1c[i] = bih1[i] + bhh1[i]; }
  #pragma unroll
  for (int i = 0; i < 5; ++i)
    #pragma unroll
    for (int j = 0; j < 5; ++j) whh1[i][j] = Whh1[i * 5 + j];

  // ---- resident A fragments: permuted Wcat (112 x 128)
  // Output position (t7, r=col) computes hidden unit in slot L = slot(t7, col).
  // Column k is the B slot index: k<100 -> Whh2[L][k]; k=104..108 -> Wih2[L][k-104];
  // k=109 -> bih2[L]+bhh2[L]; all other k (and L>=100) -> 0.
  // A layout (16-bit 16x32): lane = row M (mod 16); element e -> k = 32c + 16*hs + e
  V16H A[7][4];
  #pragma unroll
  for (int t7 = 0; t7 < 7; ++t7) {
    const int L = 32 * (t7 >> 1) + 16 * (col >> 3) + 8 * (t7 & 1) + (col & 7);
    #pragma unroll
    for (int c4 = 0; c4 < 4; ++c4) {
      #pragma unroll
      for (int e = 0; e < 16; ++e) {
        const int k = 32 * c4 + 16 * hs + e;
        float w = 0.f;
        if (L < H2) {
          if (k < H2)                     w = Whh2[L * H2 + k];
          else if (k >= 104 && k <= 108)  w = Wih2[L * 5 + (k - 104)];
          else if (k == 109)              w = bih2[L] + bhh2[L];
        }
        A[t7][c4].h[e] = (_Float16)w;
      }
    }
  }

  // ---- recurrent state: h2 in f32 D-layout (7 tiles), h1 scalar per lane
  V8F acc[7];
  #pragma unroll
  for (int t7 = 0; t7 < 7; ++t7)
    #pragma unroll
    for (int j = 0; j < 8; ++j) acc[t7].f[j] = 0.f;
  float h1[5] = {0.f, 0.f, 0.f, 0.f, 0.f};

  const float* xp = x + (size_t)(base + col) * T;   // x is (B, T, 1)
  float xcur = xp[0];

  #pragma unroll 1
  for (int t = 0; t < T; ++t) {
    // --- preload next x; its wait overlaps the in-flight WMMA chain
    const float xt = xcur;
    const int tn = (t + 1 < T) ? t + 1 : t;
    xcur = xp[tn];

    // --- layer 1 (5-dim recurrence, duplicated on both lane halves)
    float s0, s1, s2, s3, s4;
    {
      float s[5];
      #pragma unroll
      for (int i = 0; i < 5; ++i) {
        s[i] = __builtin_fmaf(wih1[i], xt, b1c[i]);
        #pragma unroll
        for (int j = 0; j < 5; ++j) s[i] = __builtin_fmaf(whh1[i][j], h1[j], s[i]);
      }
      s0 = s[0]; s1 = s[1]; s2 = s[2]; s3 = s[3]; s4 = s[4];
    }
    tanh5(s0, s1, s2, s3, s4);
    h1[0] = s0; h1[1] = s1; h1[2] = s2; h1[3] = s3; h1[4] = s4;

    // --- build B fragments: purely lane-local repack thanks to the row permutation.
    // B layout: element e -> K = 32c + 16*hs + e; slot() maps exactly onto the
    // lane's own acc values: e<8 from acc[2c][e], e>=8 from acc[2c+1][e-8].
    V16H Bf[4];
    #pragma unroll
    for (int c4 = 0; c4 < 3; ++c4) {
      #pragma unroll
      for (int j = 0; j < 4; ++j) {
        Bf[c4].u[j]     = pk2(acc[2 * c4].f[2 * j],     acc[2 * c4].f[2 * j + 1]);
        Bf[c4].u[4 + j] = pk2(acc[2 * c4 + 1].f[2 * j], acc[2 * c4 + 1].f[2 * j + 1]);
      }
    }
    // chunk 3: tile 6 values (slots 96..103 / 112..119; padding rows are exact 0),
    // then the augmented inputs h1 (slots 104..108) and bias-one (slot 109) on half 0.
    Bf[3].u[0] = pk2(acc[6].f[0], acc[6].f[1]);
    Bf[3].u[1] = pk2(acc[6].f[2], acc[6].f[3]);
    Bf[3].u[2] = pk2(acc[6].f[4], acc[6].f[5]);
    Bf[3].u[3] = pk2(acc[6].f[6], acc[6].f[7]);
    Bf[3].u[4] = hs ? 0u : pk2(h1[0], h1[1]);
    Bf[3].u[5] = hs ? 0u : pk2(h1[2], h1[3]);
    Bf[3].u[6] = hs ? 0u : pk2(h1[4], 1.0f);
    Bf[3].u[7] = 0u;

    // --- 28 WMMAs: D(112x16) = Wcat(112x128) x state(128x16), f32 accumulation
    #pragma unroll
    for (int t7 = 0; t7 < 7; ++t7) {
      v8f c = {0.f, 0.f, 0.f, 0.f, 0.f, 0.f, 0.f, 0.f};
      #pragma unroll
      for (int c4 = 0; c4 < 4; ++c4)
        c = __builtin_amdgcn_wmma_f32_16x16x32_f16(
              false, A[t7][c4].v, false, Bf[c4].v, (short)0, c, false, false);
      acc[t7].v = c;
    }

    // --- pointwise tanh, one trans-op burst per tile (padding rows stay exactly 0)
    #pragma unroll
    for (int t7 = 0; t7 < 7; ++t7) tanh8(acc[t7]);
  }

  // ---- readout: relu(h2_T . W3 + b3); lane holds positions (t7, 8*hs+j) -> slot L
  float partial = 0.f;
  #pragma unroll
  for (int t7 = 0; t7 < 7; ++t7) {
    #pragma unroll
    for (int j = 0; j < 8; ++j) {
      const int L = 32 * (t7 >> 1) + 16 * hs + 8 * (t7 & 1) + j;
      if (L < H2) partial = __builtin_fmaf(W3[L], acc[t7].f[j], partial);
    }
  }
  float tot = partial + __shfl_xor(partial, 16, 32) + b3[0];
  if (hs == 0) out[base + col] = tot > 0.f ? tot : 0.f;
}

extern "C" void kernel_launch(void* const* d_in, const int* in_sizes, int n_in,
                              void* d_out, int out_size, void* d_ws, size_t ws_size,
                              hipStream_t stream) {
  const float* x    = (const float*)d_in[0];
  const float* Wih1 = (const float*)d_in[1];
  const float* Whh1 = (const float*)d_in[2];
  const float* bih1 = (const float*)d_in[3];
  const float* bhh1 = (const float*)d_in[4];
  const float* Wih2 = (const float*)d_in[5];
  const float* Whh2 = (const float*)d_in[6];
  const float* bih2 = (const float*)d_in[7];
  const float* bhh2 = (const float*)d_in[8];
  const float* W3   = (const float*)d_in[9];
  const float* b3   = (const float*)d_in[10];
  float* out = (float*)d_out;

  const int B = in_sizes[0] / 256;        // x is (B, 256, 1)
  const int waves = (B + 15) / 16;        // 16 batch rows per wave32
  const int blocks = (waves + 7) / 8;     // 256 threads = 8 waves per block
  rnn_fused<<<blocks, 256, 0, stream>>>(x, Wih1, Whh1, bih1, bhh1,
                                        Wih2, Whh2, bih2, bhh2, W3, b3, out, B);
}
